// MyBiGRU_crf_49486613184879
// MI455X (gfx1250) — matrix operand order repaired and checked by reading, not compiled, not run on hardware
//
#include <hip/hip_runtime.h>
#include <hip/hip_bf16.h>

// ---------------------------------------------------------------------------
// BiLSTM-CRF for MI455X (gfx1250): bf16 WMMA GEMMs + LDS-resident recurrent
// weights + persistent-kernel LSTM with grid barrier + scalar Viterbi.
// ---------------------------------------------------------------------------

#define BB 128
#define SS 512
#define VV 6000
#define EE 100
#define EP 128      // E padded to multiple of 32 for WMMA K
#define HH 256
#define G4 1024     // 4*H
#define TT 9

typedef __attribute__((ext_vector_type(16))) __bf16 bf16x16;
typedef __attribute__((ext_vector_type(8)))  __bf16 bf16x8;
typedef __attribute__((ext_vector_type(8)))  float  floatx8;

static __device__ inline bf16x16 combine8(bf16x8 lo, bf16x8 hi) {
    bf16x16 r;
#pragma unroll
    for (int i = 0; i < 8; ++i) { r[i] = lo[i]; r[i + 8] = hi[i]; }
    return r;
}

static __device__ inline float sigm(float x) {
    return 1.0f / (1.0f + __expf(-x));
}
static __device__ inline float tanh_fast(float x) {
    x = fminf(15.0f, fmaxf(-15.0f, x));
    float e = __expf(2.0f * x);
    return (e - 1.0f) / (e + 1.0f);
}

// ---------------------------------------------------------------------------
// 1) f32 -> bf16 convert with column padding (weights + embedding table)
// ---------------------------------------------------------------------------
__global__ void cvt_pad_bf16(__bf16* __restrict__ dst, const float* __restrict__ src,
                             int rows, int scols, int dcols) {
    int idx = blockIdx.x * blockDim.x + threadIdx.x;
    int total = rows * dcols;
    if (idx >= total) return;
    int r = idx / dcols, c = idx - r * dcols;
    float v = (c < scols) ? src[r * scols + c] : 0.0f;
    dst[idx] = (__bf16)v;
}

// ---------------------------------------------------------------------------
// 2) Embedding gather: x_bf16[b*S+s][0..127] = emb_bf16[data[b*S+s]]
//    (row 0 of emb is zero -> padding tokens give zero rows)
// ---------------------------------------------------------------------------
__global__ void gather_rows(unsigned long long* __restrict__ dst,
                            const unsigned long long* __restrict__ embq,
                            const int* __restrict__ data, int nrows) {
    int idx = blockIdx.x * blockDim.x + threadIdx.x;   // nrows * 32 u64 each
    if (idx >= nrows * 32) return;
    int r = idx >> 5, o = idx & 31;
    int tok = data[r];
    dst[idx] = embq[(size_t)tok * 32 + o];
}

// ---------------------------------------------------------------------------
// 3) Input projection: XG[s][b][n] = x[b,s,:] . W_ih[n,:] + bias[n]   (bf16 out)
//    GEMM M=65536 (b*S+s), K=128, N=1024; wave = one 16-row M block, all N.
// ---------------------------------------------------------------------------
__global__ __launch_bounds__(256)
void proj_kernel(const __bf16* __restrict__ X,    // [65536,128]
                 const __bf16* __restrict__ W,    // [1024,128]
                 const float*  __restrict__ bias, // [1024]
                 __bf16* __restrict__ XG)         // [512,128,1024]
{
    int wave = threadIdx.x >> 5, lane = threadIdx.x & 31;
    int ln = lane & 15, lh = lane >> 4;
    int m0 = (blockIdx.x * 8 + wave) * 16;

    bf16x16 afr[4];
    const __bf16* xrow = X + (size_t)(m0 + ln) * EP;
#pragma unroll
    for (int kt = 0; kt < 4; ++kt) {
        bf16x8 lo = *(const bf16x8*)(xrow + kt * 32 + 8 * lh);
        bf16x8 hi = *(const bf16x8*)(xrow + kt * 32 + 16 + 8 * lh);
        afr[kt] = combine8(lo, hi);
    }
    for (int nt = 0; nt < 64; ++nt) {
        int n0 = nt * 16;
        floatx8 acc = {};
#pragma unroll
        for (int kt = 0; kt < 4; ++kt) {
            bf16x16 bfr = combine8(*(const bf16x8*)(W + (size_t)(n0 + ln) * EP + kt * 32 + 8 * lh), *(const bf16x8*)(W + (size_t)(n0 + ln) * EP + kt * 32 + 16 + 8 * lh));
            acc = __builtin_amdgcn_wmma_f32_16x16x32_bf16(false, afr[kt], false, bfr,
                                                          (short)0, acc, false, false);
        }
        float bv = bias[n0 + ln];
#pragma unroll
        for (int j = 0; j < 8; ++j) {
            int m  = m0 + j + 8 * lh;
            int ss = m & (SS - 1), bb = m >> 9;
            XG[((size_t)ss * BB + bb) * G4 + n0 + ln] = (__bf16)(acc[j] + bv);
        }
    }
}

// ---------------------------------------------------------------------------
// Grid barrier (sense via generation counter)
// ---------------------------------------------------------------------------
static __device__ inline void gridbar(unsigned* sync, unsigned nwg) {
    __syncthreads();
    if (threadIdx.x == 0) {
        unsigned* cnt = sync;
        unsigned* gen = sync + 1;
        unsigned g = __atomic_load_n(gen, __ATOMIC_ACQUIRE);
        unsigned prev = atomicAdd(cnt, 1u);
        if (prev == nwg - 1) {
            __atomic_store_n(cnt, 0u, __ATOMIC_RELAXED);
            __threadfence();
            atomicAdd(gen, 1u);
        } else {
            while (__atomic_load_n(gen, __ATOMIC_ACQUIRE) == g)
                __builtin_amdgcn_s_sleep(1);
        }
    }
    __syncthreads();
}

// ---------------------------------------------------------------------------
// 4) LSTM recurrence, one direction. 16 WGs x 256 threads (8 waves).
//    WG w owns h-columns [16w,16w+16); its 4 gate slices (64 rows of W_hh,
//    32KB bf16) live in LDS for all 512 steps. h ping-pongs in global bf16.
//    Each wave: 16 batch rows x 16 hcols; c state in registers matching the
//    WMMA C/D layout (lane holds j=0..7 -> m=m0+j+8h, n=lane&15).
// ---------------------------------------------------------------------------
__global__ __launch_bounds__(256)
void lstm_kernel(const __bf16* __restrict__ XG,   // [512,128,1024]
                 const __bf16* __restrict__ Whh,  // [1024,256] bf16
                 __bf16* hA, __bf16* hB,          // ping-pong [128,256], hA = h0 = 0
                 __bf16* __restrict__ hist,       // [512,128,256]
                 unsigned* sync, int reverse)
{
    __shared__ __bf16 Wl[64 * HH];                // 32 KB
    int w = blockIdx.x;
    int tid = threadIdx.x;
    {   // cooperative LDS fill: local row q -> W_hh[gate*256 + 16w + (q&15)]
        const unsigned long long* src = (const unsigned long long*)Whh;
        unsigned long long* dstl = (unsigned long long*)Wl;
        for (int i = tid; i < 64 * 64; i += 256) {
            int q = i >> 6, off = i & 63;
            int grow = (q >> 4) * HH + 16 * w + (q & 15);
            dstl[i] = src[(size_t)grow * 64 + off];
        }
    }
    __syncthreads();

    int wave = tid >> 5, lane = tid & 31;
    int ln = lane & 15, lh = lane >> 4;
    int m0 = wave * 16;
    int mrow = m0 + ln;            // A-frag row for this lane
    int colg = 16 * w + ln;        // output h column for this lane

    float c[8];
#pragma unroll
    for (int j = 0; j < 8; ++j) c[j] = 0.0f;

    for (int it = 0; it < SS; ++it) {
        int s = reverse ? (SS - 1 - it) : it;
        const __bf16* hp = (it & 1) ? hB : hA;
        __bf16*       hn = (it & 1) ? hA : hB;

        // gate accumulators initialized with precomputed input projection
        const __bf16* xgb = XG + (size_t)s * BB * G4;
        floatx8 acc[4];
#pragma unroll
        for (int g = 0; g < 4; ++g) {
            int n = g * HH + colg;
#pragma unroll
            for (int j = 0; j < 8; ++j) {
                int m = m0 + j + 8 * lh;
                acc[g][j] = (float)xgb[(size_t)m * G4 + n];
            }
        }
        // hint the next step's gate inputs toward L2
        {
            int s2 = reverse ? (s - 1) : (s + 1);
            if (s2 >= 0 && s2 < SS)
                __builtin_prefetch(XG + ((size_t)s2 * BB + mrow) * G4 + colg, 0, 1);
        }

        // h_prev @ W_hh^T : K=256 in 8 WMMA slices x 4 gate tiles
        const __bf16* arow = hp + (size_t)mrow * HH;
#pragma unroll
        for (int kt = 0; kt < 8; ++kt) {
            int kk = kt * 32;
            bf16x8 lo = *(const bf16x8*)(arow + kk + 8 * lh);
            bf16x8 hi = *(const bf16x8*)(arow + kk + 16 + 8 * lh);
            bf16x16 afr = combine8(lo, hi);
#pragma unroll
            for (int g = 0; g < 4; ++g) {
                bf16x16 bfr = combine8(*(const bf16x8*)(Wl + (size_t)(g * 16 + ln) * HH + kk + 8 * lh), *(const bf16x8*)(Wl + (size_t)(g * 16 + ln) * HH + kk + 16 + 8 * lh));
                acc[g] = __builtin_amdgcn_wmma_f32_16x16x32_bf16(false, afr, false, bfr,
                                                                 (short)0, acc[g], false, false);
            }
        }

        // gates i,f,g,o (PyTorch order) -> c,h update; write h bf16
#pragma unroll
        for (int j = 0; j < 8; ++j) {
            int m = m0 + j + 8 * lh;
            float ig = sigm(acc[0][j]);
            float fg = sigm(acc[1][j]);
            float gg = tanh_fast(acc[2][j]);
            float og = sigm(acc[3][j]);
            c[j] = fg * c[j] + ig * gg;
            __bf16 hv = (__bf16)(og * tanh_fast(c[j]));
            hn[(size_t)m * HH + colg] = hv;
            hist[((size_t)s * BB + m) * HH + colg] = hv;
        }
        __threadfence();
        gridbar(sync, 16);
    }
}

// ---------------------------------------------------------------------------
// 5) Emission MLP: em[b][s][t] = [hf;hb] . mlp_W[t,:] + mlp_b[t]
//    one wave per (b,s); lanes split the 512-wide dot; butterfly reduce.
// ---------------------------------------------------------------------------
__global__ __launch_bounds__(256)
void mlp_kernel(const __bf16* __restrict__ hf, const __bf16* __restrict__ hb,
                const float* __restrict__ W, const float* __restrict__ bias,
                float* __restrict__ em)
{
    int wid  = (blockIdx.x * 256 + threadIdx.x) >> 5;   // 0..65535 = b*S+s
    int lane = threadIdx.x & 31;
    int b = wid >> 9, s = wid & (SS - 1);

    bf16x8 vf = *(const bf16x8*)(hf + ((size_t)s * BB + b) * HH + lane * 8);
    bf16x8 vb = *(const bf16x8*)(hb + ((size_t)s * BB + b) * HH + lane * 8);
    float xf[8], xb[8];
#pragma unroll
    for (int j = 0; j < 8; ++j) { xf[j] = (float)vf[j]; xb[j] = (float)vb[j]; }

    float out[TT];
#pragma unroll
    for (int t = 0; t < TT; ++t) {
        const float* wr = W + (size_t)t * (2 * HH) + lane * 8;
        float a = 0.0f;
#pragma unroll
        for (int j = 0; j < 8; ++j) a += xf[j] * wr[j] + xb[j] * wr[HH + j];
#pragma unroll
        for (int off = 16; off > 0; off >>= 1) a += __shfl_xor(a, off, 32);
        out[t] = a;
    }
    if (lane == 0) {
        float* dst = em + ((size_t)b * SS + s) * TT;
#pragma unroll
        for (int t = 0; t < TT; ++t) dst[t] = out[t] + bias[t];
    }
}

// ---------------------------------------------------------------------------
// 6) Viterbi (T=9): one thread per batch; 4-bit backpointers packed in u64.
//    mask recomputed as (data != 0), identical to the reference construction.
// ---------------------------------------------------------------------------
__global__ __launch_bounds__(128)
void viterbi_kernel(const float* __restrict__ em, const int* __restrict__ data,
                    const float* __restrict__ startt, const float* __restrict__ trans,
                    const float* __restrict__ endt,
                    unsigned long long* __restrict__ bp,
                    int* __restrict__ paths, float* __restrict__ score_out)
{
    __shared__ float tr[TT * TT];
    if (threadIdx.x < TT * TT) tr[threadIdx.x] = trans[threadIdx.x];
    __syncthreads();

    int b = threadIdx.x;
    float sc[TT];
#pragma unroll
    for (int t = 0; t < TT; ++t) sc[t] = startt[t] + em[(size_t)b * SS * TT + t];

    for (int s = 1; s < SS; ++s) {
        const float* e = em + ((size_t)b * SS + s) * TT;
        bool m = data[b * SS + s] != 0;
        float ns[TT];
        unsigned long long wbits = 0;
#pragma unroll
        for (int tc = 0; tc < TT; ++tc) {
            float best = -3.4e38f; int arg = 0;
#pragma unroll
            for (int tp = 0; tp < TT; ++tp) {
                float v = sc[tp] + tr[tp * TT + tc];
                if (v > best) { best = v; arg = tp; }
            }
            ns[tc] = best + e[tc];
            wbits |= (unsigned long long)arg << (4 * tc);
        }
        bp[(size_t)(s - 1) * BB + b] = wbits;
        if (m) {
#pragma unroll
            for (int t = 0; t < TT; ++t) sc[t] = ns[t];
        }
    }
    float best = -3.4e38f; int tag = 0;
#pragma unroll
    for (int t = 0; t < TT; ++t) {
        float v = sc[t] + endt[t];
        if (v > best) { best = v; tag = t; }
    }
    score_out[b] = best;
    paths[b * SS + (SS - 1)] = (data[b * SS + SS - 1] != 0) ? tag : 0;
    for (int s = SS - 2; s >= 0; --s) {
        bool mn = data[b * SS + s + 1] != 0;
        unsigned long long wbits = bp[(size_t)s * BB + b];
        int prev = (int)((wbits >> (4 * tag)) & 15ull);
        tag = mn ? prev : tag;
        paths[b * SS + s] = (data[b * SS + s] != 0) ? tag : 0;
    }
}

// ---------------------------------------------------------------------------
// Host driver
// ---------------------------------------------------------------------------
extern "C" void kernel_launch(void* const* d_in, const int* in_sizes, int n_in,
                              void* d_out, int out_size, void* d_ws, size_t ws_size,
                              hipStream_t stream) {
    const int*   data  = (const int*)  d_in[0];
    // d_in[1] = mask (unused; reconstructed from data != 0)
    const float* emb   = (const float*)d_in[2];
    const float* Wih_f = (const float*)d_in[3];
    const float* Whh_f = (const float*)d_in[4];
    const float* b_f   = (const float*)d_in[5];
    const float* Wih_b = (const float*)d_in[6];
    const float* Whh_b = (const float*)d_in[7];
    const float* b_b   = (const float*)d_in[8];
    const float* mlpW  = (const float*)d_in[9];
    const float* mlpb  = (const float*)d_in[10];
    const float* stt   = (const float*)d_in[11];
    const float* trans = (const float*)d_in[12];
    const float* endt  = (const float*)d_in[13];

    char* ws = (char*)d_ws;
    size_t off = 0;
    auto alloc = [&](size_t bytes) -> void* {
        void* p = ws + off;
        off += (bytes + 255) & ~(size_t)255;
        return p;
    };
    __bf16* embq   = (__bf16*)alloc((size_t)VV * EP * 2);
    __bf16* wihfq  = (__bf16*)alloc((size_t)G4 * EP * 2);
    __bf16* wihbq  = (__bf16*)alloc((size_t)G4 * EP * 2);
    __bf16* whhfq  = (__bf16*)alloc((size_t)G4 * HH * 2);
    __bf16* whhbq  = (__bf16*)alloc((size_t)G4 * HH * 2);
    __bf16* xq     = (__bf16*)alloc((size_t)BB * SS * EP * 2);
    __bf16* xg_f   = (__bf16*)alloc((size_t)SS * BB * G4 * 2);
    __bf16* xg_b   = (__bf16*)alloc((size_t)SS * BB * G4 * 2);
    __bf16* hist_f = (__bf16*)alloc((size_t)SS * BB * HH * 2);
    __bf16* hist_b = (__bf16*)alloc((size_t)SS * BB * HH * 2);
    __bf16* hA     = (__bf16*)alloc((size_t)BB * HH * 2);
    __bf16* hB     = (__bf16*)alloc((size_t)BB * HH * 2);
    float*  emis   = (float*) alloc((size_t)BB * SS * TT * 4);
    unsigned long long* bp = (unsigned long long*)alloc((size_t)(SS - 1) * BB * 8);
    unsigned* sync = (unsigned*)alloc(256);

    int* paths   = (int*)d_out;
    float* score = (float*)d_out + (size_t)BB * SS;

    // 1) convert weights / embedding table to bf16 (pad K to 128)
    {
        int n;
        n = VV * EP; cvt_pad_bf16<<<(n + 255) / 256, 256, 0, stream>>>(embq,  emb,   VV, EE, EP);
        n = G4 * EP; cvt_pad_bf16<<<(n + 255) / 256, 256, 0, stream>>>(wihfq, Wih_f, G4, EE, EP);
        n = G4 * EP; cvt_pad_bf16<<<(n + 255) / 256, 256, 0, stream>>>(wihbq, Wih_b, G4, EE, EP);
        n = G4 * HH; cvt_pad_bf16<<<(n + 255) / 256, 256, 0, stream>>>(whhfq, Whh_f, G4, HH, HH);
        n = G4 * HH; cvt_pad_bf16<<<(n + 255) / 256, 256, 0, stream>>>(whhbq, Whh_b, G4, HH, HH);
    }
    // 2) embedding gather
    {
        int nthr = BB * SS * 32;
        gather_rows<<<nthr / 256, 256, 0, stream>>>((unsigned long long*)xq,
                                                    (const unsigned long long*)embq,
                                                    data, BB * SS);
    }
    // 3) input projections (WMMA GEMM), both directions
    proj_kernel<<<512, 256, 0, stream>>>(xq, wihfq, b_f, xg_f);
    proj_kernel<<<512, 256, 0, stream>>>(xq, wihbq, b_b, xg_b);

    // 4) recurrences (persistent kernel, grid barrier per step)
    hipMemsetAsync(hA, 0, (size_t)BB * HH * 2, stream);
    hipMemsetAsync(hB, 0, (size_t)BB * HH * 2, stream);
    hipMemsetAsync(sync, 0, 8, stream);
    lstm_kernel<<<16, 256, 0, stream>>>(xg_f, whhfq, hA, hB, hist_f, sync, 0);

    hipMemsetAsync(hA, 0, (size_t)BB * HH * 2, stream);
    hipMemsetAsync(hB, 0, (size_t)BB * HH * 2, stream);
    hipMemsetAsync(sync, 0, 8, stream);
    lstm_kernel<<<16, 256, 0, stream>>>(xg_b, whhbq, hA, hB, hist_b, sync, 1);

    // 5) emissions
    mlp_kernel<<<(BB * SS * 32) / 256, 256, 0, stream>>>(hist_f, hist_b, mlpW, mlpb, emis);

    // 6) viterbi decode
    viterbi_kernel<<<1, 128, 0, stream>>>(emis, data, stt, trans, endt, bp, paths, score);
}
